// GCN_15625091022885
// MI455X (gfx1250) — compile-verified, matrix operand order checked by reading, not run on hardware
//
#include <hip/hip_runtime.h>
#include <hip/hip_bf16.h>

// ---------------------------------------------------------------------------
// CDNA5 (gfx1250, wave32) WMMA types: A/B = 16 bf16/lane (8 VGPRs),
// C/D = 8 f32/lane (8 VGPRs).
// ---------------------------------------------------------------------------
typedef __attribute__((ext_vector_type(16))) __bf16 bf16x16;
typedef __attribute__((ext_vector_type(8)))  __bf16 bf16x8;
typedef __attribute__((ext_vector_type(4)))  __bf16 bf16x4;
typedef __attribute__((ext_vector_type(8)))  float  f32x8;
typedef __attribute__((ext_vector_type(4)))  float  f32x4;

__device__ __forceinline__ __bf16 f2bf(float f) {
    unsigned u = __builtin_bit_cast(unsigned, f);
    unsigned r = (u + 0x7FFFu + ((u >> 16) & 1u)) >> 16;   // round-to-nearest-even
    return __builtin_bit_cast(__bf16, (unsigned short)r);
}
__device__ __forceinline__ float bf2f(__bf16 b) {
    unsigned u = (unsigned)__builtin_bit_cast(unsigned short, b) << 16;
    return __builtin_bit_cast(float, u);
}

// ---------------------------------------------------------------------------
// Elementwise f32 -> bf16 (8 elems/thread).
// ---------------------------------------------------------------------------
__global__ void __launch_bounds__(256) cvt_bf16_kernel(
        const float* __restrict__ src, __bf16* __restrict__ dst, long long n) {
    const long long i = ((long long)blockIdx.x * 256 + threadIdx.x) * 8;
    if (i >= n) return;
    f32x4 v0 = *(const f32x4*)(src + i);
    f32x4 v1 = *(const f32x4*)(src + i + 4);
    bf16x8 r;
#pragma unroll
    for (int j = 0; j < 4; ++j) { r[j] = f2bf(v0[j]); r[4 + j] = f2bf(v1[j]); }
    *(bf16x8*)(dst + i) = r;
}

// dst[i] = bf16(relu(src[i] + bias[i & (F-1)])), F power of two, 4 elems/thread.
__global__ void __launch_bounds__(256) bias_relu_cvt_kernel(
        const float* __restrict__ src, const float* __restrict__ bias,
        __bf16* __restrict__ dst, long long n, int F) {
    const long long i = ((long long)blockIdx.x * 256 + threadIdx.x) * 4;
    if (i >= n) return;
    const int fb = (int)(i & (long long)(F - 1));
    f32x4 v = *(const f32x4*)(src + i);
    f32x4 bz = *(const f32x4*)(bias + fb);
    bf16x4 r;
#pragma unroll
    for (int j = 0; j < 4; ++j) r[j] = f2bf(fmaxf(v[j] + bz[j], 0.0f));
    *(bf16x4*)(dst + i) = r;
}

// ---------------------------------------------------------------------------
// Pack W[K,Nc] (f32 row-major) into fragment-ordered bf16:
// tile (nt,kt) -> 32 lanes x 16 bf16 (32 contiguous bytes per lane), B layout:
//   lanes 0-15 : col N=n, elems 0..7=K{0..7},  8..15=K{16..23}
//   lanes16-31 : col N=n, elems 0..7=K{8..15}, 8..15=K{24..31}
// ---------------------------------------------------------------------------
__global__ void __launch_bounds__(256) pack_b_kernel(
        const float* __restrict__ W, __bf16* __restrict__ Bp, int K, int Nc) {
    const int kTiles = K >> 5;
    const int nTiles = Nc >> 4;
    const int tile = blockIdx.x * 8 + (threadIdx.x >> 5);
    if (tile >= kTiles * nTiles) return;
    const int nt = tile / kTiles;
    const int kt = tile % kTiles;
    const int lane = threadIdx.x & 31;
    const int c  = nt * 16 + (lane & 15);
    const int kh = (lane >> 4) << 3;
    const int k0 = kt * 32;
    bf16x16 r;
#pragma unroll
    for (int j = 0; j < 8; ++j) {
        r[j]     = f2bf(W[(size_t)(k0 + kh + j)      * Nc + c]);
        r[8 + j] = f2bf(W[(size_t)(k0 + 16 + kh + j) * Nc + c]);
    }
    *(bf16x16*)(Bp + ((size_t)tile * 32 + lane) * 16) = r;
}

// ---------------------------------------------------------------------------
// bf16 WMMA GEMM: C[M,Nc] = A[M,K](bf16,row-major) @ Bp(packed bf16).
// 8 waves/block; each wave owns a 16x32 C strip (2 WMMA acc tiles sharing one
// A fragment). Packed B for the block's 32 columns is staged in LDS once and
// each wave reads fragments with ds_load_b128. Inner loop:
//   2x global_load_b128 (A) + 4x ds_load_b128 (B) + 2x v_wmma.
// ---------------------------------------------------------------------------
template<bool OUT_BF16, bool BIAS_OUT>
__global__ void __launch_bounds__(256) wmma_gemm_bf16(
        const __bf16* __restrict__ A, const __bf16* __restrict__ Bp,
        const float* __restrict__ biasOut, void* __restrict__ Cout,
        int M, int K, int Nc) {
    extern __shared__ char smem[];                 // kTiles * 2048 bytes
    const int kTiles = K >> 5;
    // cooperative stage of 2 packed n-tiles (contiguous in Bp)
    {
        const char* src = (const char*)Bp + (size_t)(blockIdx.y * 2) * kTiles * 1024;
        const int bytes = kTiles * 2048;
        for (int off = threadIdx.x * 16; off < bytes; off += 256 * 16)
            *(f32x4*)(smem + off) = *(const f32x4*)(src + off);
    }
    __syncthreads();

    const int wave  = threadIdx.x >> 5;
    const int mBase = (blockIdx.x * 8 + wave) * 16;
    if (mBase >= M) return;                        // wave-uniform
    const int lane = threadIdx.x & 31;
    const int row  = mBase + (lane & 15);
    const int kh   = (lane >> 4) << 3;             // 0 or 8 (A layout, ISA 7.12.2)
    const __bf16* arow = A + (size_t)row * K + kh;
    const char* b0base = smem + (size_t)lane * 32;
    const char* b1base = b0base + (size_t)kTiles * 1024;

    f32x8 acc0 = {0.f,0.f,0.f,0.f,0.f,0.f,0.f,0.f};
    f32x8 acc1 = {0.f,0.f,0.f,0.f,0.f,0.f,0.f,0.f};
    for (int kt = 0; kt < kTiles; ++kt) {
        bf16x8 a0 = *(const bf16x8*)(arow + kt * 32);        // K{kh+0..7}
        bf16x8 a1 = *(const bf16x8*)(arow + kt * 32 + 16);   // K{16+kh+0..7}
        bf16x16 a = __builtin_shufflevector(a0, a1,
                        0,1,2,3,4,5,6,7,8,9,10,11,12,13,14,15);
        bf16x16 b0 = *(const bf16x16*)(b0base + (size_t)kt * 1024);
        bf16x16 b1 = *(const bf16x16*)(b1base + (size_t)kt * 1024);
        acc0 = __builtin_amdgcn_wmma_f32_16x16x32_bf16(false, a, false, b0,
                                                       (short)0, acc0, false, false);
        acc1 = __builtin_amdgcn_wmma_f32_16x16x32_bf16(false, a, false, b1,
                                                       (short)0, acc1, false, false);
    }

    const int col   = blockIdx.y * 32 + (lane & 15);
    const int rbase = mBase + ((lane >> 4) << 3);  // C/D layout
    if (OUT_BF16) {
        __bf16* C = (__bf16*)Cout;
#pragma unroll
        for (int i = 0; i < 8; ++i) {
            C[(size_t)(rbase + i) * Nc + col]      = f2bf(acc0[i]);
            C[(size_t)(rbase + i) * Nc + col + 16] = f2bf(acc1[i]);
        }
    } else {
        float* C = (float*)Cout;
        const float bv0 = BIAS_OUT ? biasOut[col]      : 0.0f;
        const float bv1 = BIAS_OUT ? biasOut[col + 16] : 0.0f;
#pragma unroll
        for (int i = 0; i < 8; ++i) {
            C[(size_t)(rbase + i) * Nc + col]      = acc0[i] + bv0;
            C[(size_t)(rbase + i) * Nc + col + 16] = acc1[i] + bv1;
        }
    }
}

// ---------------------------------------------------------------------------
// COO SpMM scatter: Hacc[rows[e],:] += vals[e] * Hin[cols[e],:]  (Hin bf16,
// gather is L2-resident; f32 atomics into zeroed accumulator).
// ---------------------------------------------------------------------------
template<int F>
__global__ void __launch_bounds__(256) spmm_scatter_kernel(
        const int* __restrict__ rows, const int* __restrict__ cols,
        const float* __restrict__ vals, const __bf16* __restrict__ Hin,
        float* __restrict__ Hacc, int E) {
    constexpr int TPE = F / 4;
    constexpr int EPB = 256 / TPE;
    const int g = threadIdx.x / TPE;
    const int t = threadIdx.x % TPE;
    const long long e = (long long)blockIdx.x * EPB + g;
    if (e >= E) return;
    const int   r = rows[e];
    const int   c = cols[e];
    const float v = vals[e];
    bf16x4 s = *(const bf16x4*)(Hin + (size_t)c * F + t * 4);
    float* dst = Hacc + (size_t)r * F + t * 4;
    atomicAdd(dst + 0, v * bf2f(s[0]));
    atomicAdd(dst + 1, v * bf2f(s[1]));
    atomicAdd(dst + 2, v * bf2f(s[2]));
    atomicAdd(dst + 3, v * bf2f(s[3]));
}

// ---------------------------------------------------------------------------
// log_softmax over 64 classes, one wave32 per node.
// ---------------------------------------------------------------------------
__global__ void __launch_bounds__(256) logsoftmax64_kernel(
        const float* __restrict__ zacc, const float* __restrict__ b2,
        float* __restrict__ out, int Nn) {
    const int wave = threadIdx.x >> 5;
    const int lane = threadIdx.x & 31;
    const long long node = (long long)blockIdx.x * 8 + wave;
    if (node >= Nn) return;
    const float* zr = zacc + (size_t)node * 64;
    float x0 = zr[lane]      + b2[lane];
    float x1 = zr[lane + 32] + b2[lane + 32];
    float m = fmaxf(x0, x1);
#pragma unroll
    for (int off = 16; off > 0; off >>= 1)
        m = fmaxf(m, __shfl_xor(m, off, 32));
    float s = __expf(x0 - m) + __expf(x1 - m);
#pragma unroll
    for (int off = 16; off > 0; off >>= 1)
        s += __shfl_xor(s, off, 32);
    const float lse = m + __logf(s);
    out[(size_t)node * 64 + lane]      = x0 - lse;
    out[(size_t)node * 64 + lane + 32] = x1 - lse;
}

// ---------------------------------------------------------------------------
// Orchestration. d_in: 0:x 1:vals 2:W1 3:b1 4:W2 5:b2 6:W3 7:b3 8:rows 9:cols
// ws regions (bytes):
//   R0 @ 0          102.4MB : xb (bf16 x)          -> hacc (f32, after GEMM1)
//   R1 @ 102.4MB     51.2MB : tmp1 (bf16 x@W1)     -> hb (bf16 relu(hacc+b1))
//   R2 @ 153.6MB     12.8MB : tmp2 (bf16 h@W2)
//   R3 @ 166.4MB     25.6MB : zacc (f32)
//   R4 @ 192.0MB     12.8MB : zb (bf16 relu(zacc+b2))
//   R5 @ 204.8MB      0.4MB : packed W1p/W2p/W3p
// ---------------------------------------------------------------------------
extern "C" void kernel_launch(void* const* d_in, const int* in_sizes, int n_in,
                              void* d_out, int out_size, void* d_ws, size_t ws_size,
                              hipStream_t stream) {
    (void)n_in; (void)out_size; (void)ws_size;
    const float* x    = (const float*)d_in[0];
    const float* vals = (const float*)d_in[1];
    const float* W1   = (const float*)d_in[2];
    const float* b1   = (const float*)d_in[3];
    const float* W2   = (const float*)d_in[4];
    const float* b2   = (const float*)d_in[5];
    const float* W3   = (const float*)d_in[6];
    const float* b3   = (const float*)d_in[7];
    const int*   rows = (const int*)d_in[8];
    const int*   cols = (const int*)d_in[9];

    const int Nn = in_sizes[0] / 512;     // 100000 nodes
    const int E  = in_sizes[1];           // 3.2M edges

    char* ws = (char*)d_ws;
    const size_t szR0 = (size_t)Nn * 512 * 2;          // == Nn*256*4
    const size_t szR1 = (size_t)Nn * 256 * 2;
    const size_t szR2 = (size_t)Nn * 64 * 2;
    const size_t szR3 = (size_t)Nn * 64 * 4;
    const size_t szR4 = (size_t)Nn * 64 * 2;
    __bf16* xb   = (__bf16*)(ws);                                  // R0
    float*  hacc = (float*)(ws);                                   // R0 (reuse)
    __bf16* tmp1 = (__bf16*)(ws + szR0);                           // R1
    __bf16* hb   = (__bf16*)(ws + szR0);                           // R1 (reuse)
    __bf16* tmp2 = (__bf16*)(ws + szR0 + szR1);                    // R2
    float*  zacc = (float*)(ws + szR0 + szR1 + szR2);              // R3
    __bf16* zb   = (__bf16*)(ws + szR0 + szR1 + szR2 + szR3);      // R4
    __bf16* Wp   = (__bf16*)(ws + szR0 + szR1 + szR2 + szR3 + szR4);
    __bf16* W1p  = Wp;                        // 512*256 bf16
    __bf16* W2p  = W1p + 512 * 256;           // 256*64
    __bf16* W3p  = W2p + 256 * 64;            // 64*128

    float* out1 = (float*)d_out;              // log_softmax(z)  [N,64]
    float* out2 = out1 + (size_t)Nn * 64;     // projection      [N,128]

    // --- pack weights (one-time, tiny) ---
    pack_b_kernel<<<(16 * 16 + 7) / 8, 256, 0, stream>>>(W1, W1p, 512, 256);
    pack_b_kernel<<<(8  *  4 + 7) / 8, 256, 0, stream>>>(W2, W2p, 256, 64);
    pack_b_kernel<<<(2  *  8 + 7) / 8, 256, 0, stream>>>(W3, W3p, 64, 128);

    // --- x -> bf16 ---
    {
        long long n = (long long)Nn * 512;
        cvt_bf16_kernel<<<(unsigned)((n / 8 + 255) / 256), 256, 0, stream>>>(x, xb, n);
    }
    // --- tmp1 = xb @ W1 ---
    {
        dim3 grid((Nn + 127) / 128, 256 / 32);
        wmma_gemm_bf16<true, false><<<grid, 256, 512 * 64, stream>>>(
            xb, W1p, nullptr, tmp1, Nn, 512, 256);
    }
    // --- hacc = 0 (R0 reuse; xb dead after GEMM1) ---
    hipMemsetAsync(hacc, 0, (size_t)Nn * 256 * sizeof(float), stream);
    // --- hacc += scatter(vals * tmp1[cols]) ---
    spmm_scatter_kernel<256><<<(E + 3) / 4, 256, 0, stream>>>(rows, cols, vals,
                                                              tmp1, hacc, E);
    // --- hb = bf16(relu(hacc + b1)) (overwrites tmp1) ---
    {
        long long n = (long long)Nn * 256;
        bias_relu_cvt_kernel<<<(unsigned)((n / 4 + 255) / 256), 256, 0, stream>>>(
            hacc, b1, hb, n, 256);
    }
    // --- tmp2 = hb @ W2 ---
    {
        dim3 grid((Nn + 127) / 128, 64 / 32);
        wmma_gemm_bf16<true, false><<<grid, 256, 256 * 64, stream>>>(
            hb, W2p, nullptr, tmp2, Nn, 256, 64);
    }
    // --- zacc = 0 ---
    hipMemsetAsync(zacc, 0, (size_t)Nn * 64 * sizeof(float), stream);
    // --- zacc += scatter(vals * tmp2[cols]) ---
    spmm_scatter_kernel<64><<<(E + 15) / 16, 256, 0, stream>>>(rows, cols, vals,
                                                               tmp2, zacc, E);
    // --- zb = bf16(relu(zacc + b2)) ---
    {
        long long n = (long long)Nn * 64;
        bias_relu_cvt_kernel<<<(unsigned)((n / 4 + 255) / 256), 256, 0, stream>>>(
            zacc, b2, zb, n, 64);
    }
    // --- out1 = log_softmax(zacc + b2) ---
    logsoftmax64_kernel<<<(Nn + 7) / 8, 256, 0, stream>>>(zacc, b2, out1, Nn);
    // --- out2 = zb @ W3 + b3 ---
    {
        dim3 grid((Nn + 127) / 128, 128 / 32);
        wmma_gemm_bf16<false, true><<<grid, 256, 64 * 64, stream>>>(
            zb, W3p, b3, out2, Nn, 64, 128);
    }
}
